// IG_RGCN_36429912605254
// MI455X (gfx1250) — matrix operand order repaired and checked by reading, not compiled, not run on hardware
//
#include <hip/hip_runtime.h>
#include <hip/hip_bf16.h>
#include <math.h>

// ---------------- problem constants (match reference) ----------------
#define CN0   100000
#define CN1   20000
#define CN2   4000
#define CR    3
#define CE0   160000
#define CE1   64000
#define CHID  128
#define CFEAT 256
#define CMSG  768

// ---------------- WMMA types ----------------
typedef __attribute__((ext_vector_type(16))) __bf16 v16bf;
typedef __attribute__((ext_vector_type(8)))  float  v8f;

union FragU { v16bf bf; unsigned int u[8]; };

// ---------------- scalar helpers ----------------
__device__ __forceinline__ unsigned short f2bf(float f) {
  unsigned int u = __float_as_uint(f);
  u += 0x7FFFu + ((u >> 16) & 1u);          // round-to-nearest-even
  return (unsigned short)(u >> 16);
}
__device__ __forceinline__ float bf2f(unsigned int h) { return __uint_as_float(h << 16); }
// order-preserving float->uint key (for atomic max on floats)
__device__ __forceinline__ unsigned int f2key(float f) {
  unsigned int b = __float_as_uint(f);
  return (b & 0x80000000u) ? ~b : (b | 0x80000000u);
}
__device__ __forceinline__ float key2f(unsigned int k) {
  unsigned int b = (k & 0x80000000u) ? (k & 0x7FFFFFFFu) : ~k;
  return __uint_as_float(b);
}
#define NEG_INF_KEY 0x007FFFFFu  // f2key(-inf)

// ---------------- WMMA A-fragment loader (bf16, row-major source) -------------
// A 16x32 bf16: lane holds row m=lane%16; elems 0..7 -> k = (lane/16)*8 + i,
//               elems 8..15 -> k = 16 + (lane/16)*8 + (i-8)
__device__ __forceinline__ v16bf load_a_frag(const unsigned short* __restrict__ arow,
                                             int kbase, int lane) {
  const unsigned short* p = arow + kbase + ((lane >> 4) << 3);
  FragU f;
  uint4 lo = *reinterpret_cast<const uint4*>(p);
  uint4 hi = *reinterpret_cast<const uint4*>(p + 16);
  f.u[0] = lo.x; f.u[1] = lo.y; f.u[2] = lo.z; f.u[3] = lo.w;
  f.u[4] = hi.x; f.u[5] = hi.y; f.u[6] = hi.z; f.u[7] = hi.w;
  return f.bf;
}

// ---------------- generic GEMM: D[M x 128] = act(A[M x K] * W^T + bias) --------
// blockDim = 128 (4 waves). Wave w owns M-tile (blockIdx.x*4 + w), all 8 N-tiles.
// Per K-step(32): the block async-stages the 128x32 bf16 W^T slab (8KB) into LDS
// (global_load_async_to_lds_b128 / ASYNCcnt), then each wave runs 8 WMMAs fed by
// ds_load_b128. B-fragment layout in LDS row n: 64 bytes, lane reads 16B at
// offset (lane/16)*16 and 16B at +32 (B frag: lane = column n%16, same k order as A).
// ACT: 0=none 1=relu 2=tanh. Optional fp32 and/or bf16 outputs (same ldd).
template <int ACT>
__global__ void k_gemm_n128(const unsigned short* __restrict__ A, int lda, int K,
                            int Mtiles,
                            const unsigned short* __restrict__ WT,      // [128 x K] bf16
                            const float* __restrict__ bias,             // [128]
                            float* __restrict__ Df,
                            unsigned short* __restrict__ Db, int ldd) {
  __shared__ unsigned char bslab[128 * 64];   // [n=0..127][32 bf16 = 64B]

  const int tid  = threadIdx.x;
  const int lane = tid & 31;
  const int wave = tid >> 5;

  int mtile = blockIdx.x * 4 + wave;
  if (mtile >= Mtiles) mtile = Mtiles - 1;    // clamp: redundant identical work,
                                              // keeps barriers convergent
  const int row0 = mtile * 16;
  const unsigned short* arow = A + (size_t)(row0 + (lane & 15)) * lda;

  const unsigned lds_base = (unsigned)(uintptr_t)(void*)&bslab[0]; // addr[31:0] = LDS offset
  const unsigned lds_dst  = lds_base + tid * 64;

  const v8f vzero = {0.f, 0.f, 0.f, 0.f, 0.f, 0.f, 0.f, 0.f};
  v8f acc[8];
#pragma unroll
  for (int nt = 0; nt < 8; ++nt) acc[nt] = vzero;

  for (int k = 0; k < K; k += 32) {
    // ---- async stage: thread t loads W^T row n=t, k..k+31 (64B) into LDS ----
    {
      const unsigned short* gsrc = WT + (size_t)tid * K + k;
#pragma unroll
      for (int c = 0; c < 4; ++c) {
        asm volatile("global_load_async_to_lds_b128 %0, %1, off"
                     :: "v"(lds_dst + c * 16), "v"(gsrc + c * 8)
                     : "memory");
      }
      asm volatile("s_wait_asynccnt 0" ::: "memory");
    }
    __syncthreads();

    if (k + 32 < K) __builtin_prefetch(arow + k + 32, 0, 0);  // global_prefetch_b8
    v16bf afrag = load_a_frag(arow, k, lane);

#pragma unroll
    for (int nt = 0; nt < 8; ++nt) {
      const unsigned char* bp =
          &bslab[(nt * 16 + (lane & 15)) * 64 + ((lane >> 4) << 4)];
      FragU bf;
      uint4 lo = *reinterpret_cast<const uint4*>(bp);       // ds_load_b128
      uint4 hi = *reinterpret_cast<const uint4*>(bp + 32);  // ds_load_b128
      bf.u[0] = lo.x; bf.u[1] = lo.y; bf.u[2] = lo.z; bf.u[3] = lo.w;
      bf.u[4] = hi.x; bf.u[5] = hi.y; bf.u[6] = hi.z; bf.u[7] = hi.w;
      acc[nt] = __builtin_amdgcn_wmma_f32_16x16x32_bf16(
          false, afrag, false, bf.bf, (short)0, acc[nt], false, false);
    }
    __syncthreads();   // slab consumed; safe to overwrite next iteration
  }

  const int col  = lane & 15;
  const int half = lane >> 4;
#pragma unroll
  for (int nt = 0; nt < 8; ++nt) {
    const int n = nt * 16 + col;
    const float bv = bias ? bias[n] : 0.f;
#pragma unroll
    for (int v = 0; v < 8; ++v) {
      const int m = row0 + v + 8 * half;    // C/D layout: VGPR v -> M = v + 8*(lane/16)
      float val = acc[nt][v] + bv;
      if (ACT == 1) val = fmaxf(val, 0.f);
      if (ACT == 2) val = tanhf(val);
      if (Df) Df[(size_t)m * ldd + n] = val;
      if (Db) Db[(size_t)m * ldd + n] = f2bf(val);
    }
  }
}

// ---------------- small utility kernels ----------------
__global__ void k_cvt_bf16(const float* __restrict__ s, unsigned short* __restrict__ d, long n) {
  long t = (long)blockIdx.x * blockDim.x + threadIdx.x;
  if (t < n) d[t] = f2bf(s[t]);
}

// batched transpose+convert: src[b][FI][FO] (fp32) -> dst[b][FO][FI] (bf16)
__global__ void k_cvtT(const float* __restrict__ s, unsigned short* __restrict__ d,
                       int FI, int FO) {
  long total = (long)FI * FO;
  long t = (long)blockIdx.x * blockDim.x + threadIdx.x;
  if (t >= total) return;
  int k = (int)(t / FO), n = (int)(t % FO);
  s += (size_t)blockIdx.y * total;
  d += (size_t)blockIdx.y * total;
  d[(size_t)n * FI + k] = f2bf(s[t]);
}

// fill feat[:,128:256] with gathered tss/rs embeddings (bf16)
__global__ void k_gather_embed(const float* __restrict__ tss, const float* __restrict__ rs,
                               const int* __restrict__ nid,
                               unsigned short* __restrict__ feat, int N) {
  int t = blockIdx.x * blockDim.x + threadIdx.x;
  if (t >= N * 128) return;
  int node = t >> 7, j = t & 127;
  int g = nid[node];
  float v = (j < 64) ? tss[(size_t)g * 64 + j] : rs[(size_t)g * 64 + (j - 64)];
  feat[(size_t)node * CFEAT + 128 + j] = f2bf(v);
}

__global__ void k_init_agg(float* __restrict__ sum, unsigned int* __restrict__ maxk,
                           float* __restrict__ deg, int N) {
  long t = (long)blockIdx.x * blockDim.x + threadIdx.x;
  long tot = (long)N * CFEAT;
  if (t < tot) { sum[t] = 0.f; maxk[t] = NEG_INF_KEY; }
  if (t < N)   deg[t] = 0.f;
}

// one wave per edge; lane handles 8 contiguous feature columns
__global__ void k_edge_agg(const unsigned short* __restrict__ feat,
                           const int* __restrict__ esrc, const int* __restrict__ edst, int E,
                           float* __restrict__ sum, unsigned int* __restrict__ maxk,
                           float* __restrict__ deg) {
  int e    = (blockIdx.x * blockDim.x + threadIdx.x) >> 5;
  int lane = threadIdx.x & 31;
  if (e >= E) return;
  int s = esrc[e], d = edst[e];
  const unsigned short* fp = feat + (size_t)s * CFEAT + lane * 8;
  uint4 pk = *reinterpret_cast<const uint4*>(fp);
  float*        srow = sum  + (size_t)d * CFEAT + lane * 8;
  unsigned int* mrow = maxk + (size_t)d * CFEAT + lane * 8;
  unsigned int w[4] = {pk.x, pk.y, pk.z, pk.w};
#pragma unroll
  for (int i = 0; i < 4; ++i) {
    float lo = bf2f(w[i] & 0xFFFFu);
    float hi = bf2f(w[i] >> 16);
    atomicAdd(srow + 2 * i,     lo);
    atomicAdd(srow + 2 * i + 1, hi);
    atomicMax(mrow + 2 * i,     f2key(lo));
    atomicMax(mrow + 2 * i + 1, f2key(hi));
  }
  if (lane == 0) atomicAdd(deg + d, 1.0f);
}

// msg[node] = [max | mean | sum] (bf16, 768 cols); DGL zeros max for isolated dst
__global__ void k_fin_msg(const float* __restrict__ sum, const unsigned int* __restrict__ maxk,
                          const float* __restrict__ deg, unsigned short* __restrict__ msg, int N) {
  long t = (long)blockIdx.x * blockDim.x + threadIdx.x;
  if (t >= (long)N * CFEAT) return;
  int node = (int)(t >> 8), c = (int)(t & 255);
  float dg = deg[node];
  float s  = sum[t];
  float mx = (dg > 0.f) ? key2f(maxk[t]) : 0.f;
  float mean = s / fmaxf(dg, 1.f);
  unsigned short* row = msg + (size_t)node * CMSG;
  row[c]        = f2bf(mx);
  row[256 + c]  = f2bf(mean);
  row[512 + c]  = f2bf(s);
}

__global__ void k_zerof(float* p, int n) {
  int t = blockIdx.x * blockDim.x + threadIdx.x;
  if (t < n) p[t] = 0.f;
}

// per-node dot(tanh_row, w2), wave-reduced, accumulated into wsum[r]
__global__ void k_attn_reduce(const float* __restrict__ ts, const float* __restrict__ w2,
                              float* __restrict__ acc_out, int N) {
  int node = blockIdx.x * blockDim.x + threadIdx.x;
  float a = 0.f;
  if (node < N) {
    const float* row = ts + (size_t)node * CHID;
    for (int j = 0; j < CHID; ++j) a += row[j] * w2[j];
  }
  for (int off = 16; off > 0; off >>= 1) a += __shfl_down(a, off, 32);
  if ((threadIdx.x & 31) == 0) atomicAdd(acc_out, a);
}

__global__ void k_beta(float* __restrict__ ws, float invN) {
  float w0 = ws[0] * invN, w1 = ws[1] * invN, w2 = ws[2] * invN;
  float mx = fmaxf(w0, fmaxf(w1, w2));
  float e0 = expf(w0 - mx), e1 = expf(w1 - mx), e2 = expf(w2 - mx);
  float s = e0 + e1 + e2;
  ws[4] = e0 / s; ws[5] = e1 / s; ws[6] = e2 / s;
}

template <bool RELU, bool WFEAT>
__global__ void k_combine(const float* __restrict__ z, const float* __restrict__ beta,
                          float* __restrict__ h, unsigned short* __restrict__ feat, int N) {
  int t = blockIdx.x * blockDim.x + threadIdx.x;
  if (t >= N * CHID) return;
  size_t st = (size_t)N * CHID;
  float v = beta[0] * z[t] + beta[1] * z[t + st] + beta[2] * z[t + 2 * st];
  if (RELU) v = fmaxf(v, 0.f);
  h[t] = v;
  if (WFEAT) {
    int node = t >> 7, c = t & 127;
    feat[(size_t)node * CFEAT + c] = f2bf(v);
  }
}

__global__ void k_predict(const float* __restrict__ h, const float* __restrict__ pw,
                          const float* __restrict__ pb, float* __restrict__ out, int N) {
  int node = blockIdx.x * blockDim.x + threadIdx.x;
  if (node >= N) return;
  const float* row = h + (size_t)node * CHID;
  float a = pb[0];
  for (int j = 0; j < CHID; ++j) a += row[j] * pw[j];
  out[node] = 1.f / (1.f + expf(-a));
}

// ---------------- workspace layout (bytes) ----------------
constexpr size_t AL(size_t x) { return (x + 255) & ~(size_t)255; }
constexpr size_t O_FEAT0 = 0;                                          // N0*256 bf16
constexpr size_t O_XU    = AL(O_FEAT0 + (size_t)CN0 * CFEAT * 2);      // N0*128 bf16
constexpr size_t O_EWT   = AL(O_XU    + (size_t)CN0 * CHID * 2);       // 128*128 bf16
constexpr size_t O_AWT   = AL(O_EWT   + 128 * 128 * 2);
constexpr size_t O_L1F1T = AL(O_AWT   + 128 * 128 * 2);                // 3*128*256
constexpr size_t O_L1F2T = AL(O_L1F1T + (size_t)CR * 128 * 256 * 2);   // 3*128*768
constexpr size_t O_L1F3T = AL(O_L1F2T + (size_t)CR * 128 * 768 * 2);
constexpr size_t O_L2F1T = AL(O_L1F3T + (size_t)CR * 128 * 256 * 2);
constexpr size_t O_L2F2T = AL(O_L2F1T + (size_t)CR * 128 * 256 * 2);
constexpr size_t O_L2F3T = AL(O_L2F2T + (size_t)CR * 128 * 768 * 2);
constexpr size_t O_SUM   = AL(O_L2F3T + (size_t)CR * 128 * 256 * 2);   // N1*256 f32
constexpr size_t O_MAXK  = AL(O_SUM   + (size_t)CN1 * CFEAT * 4);
constexpr size_t O_DEG   = AL(O_MAXK  + (size_t)CN1 * CFEAT * 4);      // N1 f32
constexpr size_t O_MSG   = AL(O_DEG   + (size_t)CN1 * 4);              // N1*768 bf16
constexpr size_t O_INP   = AL(O_MSG   + (size_t)CN1 * CMSG * 2);       // N1*256 bf16
constexpr size_t O_Z     = AL(O_INP   + (size_t)CN1 * CFEAT * 2);      // 3*N1*128 f32
constexpr size_t O_ZBF   = AL(O_Z     + (size_t)CR * CN1 * CHID * 4);  // 3*N1*128 bf16
constexpr size_t O_TANH  = AL(O_ZBF   + (size_t)CR * CN1 * CHID * 2);  // N1*128 f32
constexpr size_t O_H     = AL(O_TANH  + (size_t)CN1 * CHID * 4);       // N1*128 f32
constexpr size_t O_FEAT1 = AL(O_H     + (size_t)CN1 * CHID * 4);       // N1*256 bf16
constexpr size_t O_WSUM  = AL(O_FEAT1 + (size_t)CN1 * CFEAT * 2);      // 16 f32

static inline int cdiv(long a, long b) { return (int)((a + b - 1) / b); }

// ---------------- one IGConv layer (3 relations) ----------------
static void run_igconv_layer(const unsigned short* feat, int Ndst,
                             const int* esrc, const int* edst, int E,
                             const unsigned short* f1T, const float* f1b,
                             const unsigned short* f2T, const float* f2b,
                             const unsigned short* f3T, const float* f3b,
                             float* sum, unsigned int* maxk, float* deg,
                             unsigned short* msg, unsigned short* inp,
                             float* z, unsigned short* zbf, hipStream_t stream) {
  const long tot = (long)Ndst * CFEAT;
  const int mt = Ndst / 16;
  for (int r = 0; r < CR; ++r) {
    k_init_agg<<<cdiv(tot, 256), 256, 0, stream>>>(sum, maxk, deg, Ndst);
    k_edge_agg<<<cdiv((long)E * 32, 256), 256, 0, stream>>>(
        feat, esrc + (size_t)r * E, edst + (size_t)r * E, E, sum, maxk, deg);
    k_fin_msg<<<cdiv(tot, 256), 256, 0, stream>>>(sum, maxk, deg, msg, Ndst);
    // fc2(msg) -> relu -> inp[:, 0:128]
    k_gemm_n128<1><<<cdiv(mt, 4), 128, 0, stream>>>(
        msg, CMSG, CMSG, mt, f2T + (size_t)r * 128 * CMSG, f2b + r * 128,
        nullptr, inp, CFEAT);
    // fc1(dst_feat) -> relu -> inp[:, 128:256]
    k_gemm_n128<1><<<cdiv(mt, 4), 128, 0, stream>>>(
        feat, CFEAT, CFEAT, mt, f1T + (size_t)r * 128 * CFEAT, f1b + r * 128,
        nullptr, inp + 128, CFEAT);
    // fc3(inp) -> z[r] (f32 + bf16)
    k_gemm_n128<0><<<cdiv(mt, 4), 128, 0, stream>>>(
        inp, CFEAT, CFEAT, mt, f3T + (size_t)r * 128 * CFEAT, f3b + r * 128,
        z + (size_t)r * Ndst * CHID, zbf + (size_t)r * Ndst * CHID, CHID);
  }
}

static void run_attn(const unsigned short* zbf, int N,
                     const unsigned short* awT, const float* ab1, const float* aw2,
                     float* tanh_s, float* wsum, hipStream_t stream) {
  k_zerof<<<1, 32, 0, stream>>>(wsum, 8);
  const int mt = N / 16;
  for (int r = 0; r < CR; ++r) {
    k_gemm_n128<2><<<cdiv(mt, 4), 128, 0, stream>>>(
        zbf + (size_t)r * N * CHID, CHID, CHID, mt, awT, ab1, tanh_s, nullptr, CHID);
    k_attn_reduce<<<cdiv(N, 256), 256, 0, stream>>>(tanh_s, aw2, wsum + r, N);
  }
  k_beta<<<1, 1, 0, stream>>>(wsum, 1.0f / (float)N);
}

// ---------------- entry point ----------------
extern "C" void kernel_launch(void* const* d_in, const int* in_sizes, int n_in,
                              void* d_out, int out_size, void* d_ws, size_t ws_size,
                              hipStream_t stream) {
  (void)in_sizes; (void)n_in; (void)out_size; (void)ws_size;

  const float* x_user = (const float*)d_in[0];
  const float* tss    = (const float*)d_in[1];
  const float* rsE    = (const float*)d_in[2];
  const int*   nid0   = (const int*)d_in[3];
  const int*   nid1   = (const int*)d_in[4];
  const int*   e0s    = (const int*)d_in[5];
  const int*   e0d    = (const int*)d_in[6];
  const int*   e1s    = (const int*)d_in[7];
  const int*   e1d    = (const int*)d_in[8];
  const float* embw   = (const float*)d_in[9];
  const float* embb   = (const float*)d_in[10];
  const float* l1f1w  = (const float*)d_in[11]; const float* l1f1b = (const float*)d_in[12];
  const float* l1f2w  = (const float*)d_in[13]; const float* l1f2b = (const float*)d_in[14];
  const float* l1f3w  = (const float*)d_in[15]; const float* l1f3b = (const float*)d_in[16];
  const float* l2f1w  = (const float*)d_in[17]; const float* l2f1b = (const float*)d_in[18];
  const float* l2f2w  = (const float*)d_in[19]; const float* l2f2b = (const float*)d_in[20];
  const float* l2f3w  = (const float*)d_in[21]; const float* l2f3b = (const float*)d_in[22];
  const float* aw1    = (const float*)d_in[23]; const float* ab1   = (const float*)d_in[24];
  const float* aw2    = (const float*)d_in[25];
  const float* pw     = (const float*)d_in[26]; const float* pb    = (const float*)d_in[27];
  float* out = (float*)d_out;

  char* ws = (char*)d_ws;
  unsigned short* feat0 = (unsigned short*)(ws + O_FEAT0);
  unsigned short* xu    = (unsigned short*)(ws + O_XU);
  unsigned short* ewT   = (unsigned short*)(ws + O_EWT);
  unsigned short* awT   = (unsigned short*)(ws + O_AWT);
  unsigned short* f1T1  = (unsigned short*)(ws + O_L1F1T);
  unsigned short* f2T1  = (unsigned short*)(ws + O_L1F2T);
  unsigned short* f3T1  = (unsigned short*)(ws + O_L1F3T);
  unsigned short* f1T2  = (unsigned short*)(ws + O_L2F1T);
  unsigned short* f2T2  = (unsigned short*)(ws + O_L2F2T);
  unsigned short* f3T2  = (unsigned short*)(ws + O_L2F3T);
  float*          sum   = (float*)(ws + O_SUM);
  unsigned int*   maxk  = (unsigned int*)(ws + O_MAXK);
  float*          deg   = (float*)(ws + O_DEG);
  unsigned short* msg   = (unsigned short*)(ws + O_MSG);
  unsigned short* inp   = (unsigned short*)(ws + O_INP);
  float*          z     = (float*)(ws + O_Z);
  unsigned short* zbf   = (unsigned short*)(ws + O_ZBF);
  float*          tanhs = (float*)(ws + O_TANH);
  float*          hbuf  = (float*)(ws + O_H);
  unsigned short* feat1 = (unsigned short*)(ws + O_FEAT1);
  float*          wsum  = (float*)(ws + O_WSUM);

  // ---- prep: convert inputs / transpose weights to bf16 ----
  k_cvt_bf16<<<cdiv((long)CN0 * CHID, 256), 256, 0, stream>>>(x_user, xu, (long)CN0 * CHID);
  k_cvtT<<<dim3(cdiv(128 * 128, 256), 1), 256, 0, stream>>>(embw, ewT, 128, 128);
  k_cvtT<<<dim3(cdiv(128 * 128, 256), 1), 256, 0, stream>>>(aw1,  awT, 128, 128);
  k_cvtT<<<dim3(cdiv(256 * 128, 256), CR), 256, 0, stream>>>(l1f1w, f1T1, 256, 128);
  k_cvtT<<<dim3(cdiv(768 * 128, 256), CR), 256, 0, stream>>>(l1f2w, f2T1, 768, 128);
  k_cvtT<<<dim3(cdiv(256 * 128, 256), CR), 256, 0, stream>>>(l1f3w, f3T1, 256, 128);
  k_cvtT<<<dim3(cdiv(256 * 128, 256), CR), 256, 0, stream>>>(l2f1w, f1T2, 256, 128);
  k_cvtT<<<dim3(cdiv(768 * 128, 256), CR), 256, 0, stream>>>(l2f2w, f2T2, 768, 128);
  k_cvtT<<<dim3(cdiv(256 * 128, 256), CR), 256, 0, stream>>>(l2f3w, f3T2, 256, 128);

  // ---- feat0 = [x_user @ embed_w + b | tss[nid0] | rs[nid0]]  (bf16) ----
  k_gemm_n128<0><<<cdiv(CN0 / 16, 4), 128, 0, stream>>>(
      xu, CHID, CHID, CN0 / 16, ewT, embb, nullptr, feat0, CFEAT);
  k_gather_embed<<<cdiv((long)CN0 * CHID, 256), 256, 0, stream>>>(tss, rsE, nid0, feat0, CN0);

  // ---- layer 1 ----
  run_igconv_layer(feat0, CN1, e0s, e0d, CE0, f1T1, l1f1b, f2T1, l1f2b, f3T1, l1f3b,
                   sum, maxk, deg, msg, inp, z, zbf, stream);
  run_attn(zbf, CN1, awT, ab1, aw2, tanhs, wsum, stream);
  k_combine<true, true><<<cdiv((long)CN1 * CHID, 256), 256, 0, stream>>>(
      z, wsum + 4, hbuf, feat1, CN1);
  k_gather_embed<<<cdiv((long)CN1 * CHID, 256), 256, 0, stream>>>(tss, rsE, nid1, feat1, CN1);

  // ---- layer 2 ----
  run_igconv_layer(feat1, CN2, e1s, e1d, CE1, f1T2, l2f1b, f2T2, l2f2b, f3T2, l2f3b,
                   sum, maxk, deg, msg, inp, z, zbf, stream);
  run_attn(zbf, CN2, awT, ab1, aw2, tanhs, wsum, stream);
  k_combine<false, false><<<cdiv((long)CN2 * CHID, 256), 256, 0, stream>>>(
      z, wsum + 4, hbuf, nullptr, CN2);

  // ---- prediction head ----
  k_predict<<<cdiv(CN2, 256), 256, 0, stream>>>(hbuf, pw, pb, out, CN2);
}
